// StudentModel_90572270338534
// MI455X (gfx1250) — matrix operand order repaired and checked by reading, not compiled, not run on hardware
//
#include <hip/hip_runtime.h>
#include <stdint.h>

// ---------------------------------------------------------------------------
// Types for WMMA fragments (gfx1250, wave32)
// ---------------------------------------------------------------------------
typedef __attribute__((ext_vector_type(16))) __bf16 v16bf;
typedef __attribute__((ext_vector_type(8)))  float  v8f;

struct __align__(16) U128 { unsigned x, y, z, w; };
union FragU { U128 q[2]; v16bf v; };

__device__ __forceinline__ unsigned short f2bf(float f) {
  unsigned u = __float_as_uint(f);
  u += 0x7FFFu + ((u >> 16) & 1u);            // round-to-nearest-even
  return (unsigned short)(u >> 16);
}
__device__ __forceinline__ unsigned ordU(float f) {   // order-preserving float->uint
  unsigned u = __float_as_uint(f);
  return (u & 0x80000000u) ? ~u : (u | 0x80000000u);
}
__device__ __forceinline__ float unordU(unsigned u) {
  unsigned v = (u & 0x80000000u) ? (u & 0x7FFFFFFFu) : ~u;
  return __uint_as_float(v);
}

// ---------------------------------------------------------------------------
// float -> bf16 conversion with K padding: src [rows,K] -> dst [rows,Kpad]
// ---------------------------------------------------------------------------
__global__ void gat_cvt_bf16(const float* __restrict__ src, unsigned short* __restrict__ dst,
                             int rows, int K, int Kpad) {
  long idx = (long)blockIdx.x * blockDim.x + threadIdx.x;
  long tot = (long)rows * Kpad;
  if (idx >= tot) return;
  int r = (int)(idx / Kpad), c = (int)(idx % Kpad);
  float v = (c < K) ? src[(size_t)r * K + c] : 0.0f;
  dst[idx] = f2bf(v);
}

// ---------------------------------------------------------------------------
// WMMA bf16 GEMM:  C[M,Ncols] = A[M,Kpad] * B[Ncols,Kpad]^T   (f32 accumulate)
// Block: 256 threads (8 waves), tile 128(M) x 128(N), K-step 32.
// Waves: 4 along M x 2 along N; each wave -> 32x64 patch = 8 accumulators,
// 8 WMMAs per K-step from 12 ds_load_b128 (1.5 loads/WMMA).
// Double-buffered LDS, one barrier per K-step. OOB rows CLAMPED (results
// discarded by store guards) so loads stay plain global_load_b128.
// ---------------------------------------------------------------------------
__launch_bounds__(256)
__global__ void gat_gemm_bf16(const unsigned short* __restrict__ A,
                              const unsigned short* __restrict__ B,
                              float* __restrict__ Cout,
                              int M, int Ncols, int K, int ldC) {
  __shared__ __align__(16) unsigned short sA[2][128 * 40];  // stride 40 halves
  __shared__ __align__(16) unsigned short sB[2][128 * 40];

  const int t       = threadIdx.x;
  const int lane    = t & 31;
  const int wave    = t >> 5;
  const int laneLow = lane & 15;
  const int laneHi  = lane >> 4;           // 0 or 1
  const int mBlk    = blockIdx.y * 128;
  const int nBlk    = blockIdx.x * 128;
  const int mW      = (wave & 3) * 32;     // 4 waves along M
  const int nW      = (wave >> 2) * 64;    // 2 waves along N

  const int ar = t >> 2;                   // 0..63
  const int ac = (t & 3) * 8;              // half offset: 0,8,16,24

  // clamped source rows (duplicate last row; results discarded by store guards)
  int ga0 = mBlk + ar;       if (ga0 >= M) ga0 = M - 1;
  int ga1 = mBlk + ar + 64;  if (ga1 >= M) ga1 = M - 1;
  int gb0 = nBlk + ar;       if (gb0 >= Ncols) gb0 = Ncols - 1;
  int gb1 = nBlk + ar + 64;  if (gb1 >= Ncols) gb1 = Ncols - 1;

  const unsigned char* pA0 = (const unsigned char*)A + ((size_t)ga0 * K + ac) * 2;
  const unsigned char* pA1 = (const unsigned char*)A + ((size_t)ga1 * K + ac) * 2;
  const unsigned char* pB0 = (const unsigned char*)B + ((size_t)gb0 * K + ac) * 2;
  const unsigned char* pB1 = (const unsigned char*)B + ((size_t)gb1 * K + ac) * 2;

  v8f acc[2][4] = {};
  const int nk = K >> 5;

  // prologue: tile 0 -> buffer 0
  {
    U128 a0 = *(const U128*)pA0;
    U128 a1 = *(const U128*)pA1;
    U128 b0 = *(const U128*)pB0;
    U128 b1 = *(const U128*)pB1;
    *(U128*)(&sA[0][0] + ar * 40 + ac)        = a0;
    *(U128*)(&sA[0][0] + (ar + 64) * 40 + ac) = a1;
    *(U128*)(&sB[0][0] + ar * 40 + ac)        = b0;
    *(U128*)(&sB[0][0] + (ar + 64) * 40 + ac) = b1;
  }

  for (int it = 0; it < nk; ++it) {
    __syncthreads();                         // buffer it&1 is ready for all waves
    const unsigned short* rA = &sA[it & 1][0];
    const unsigned short* rB = &sB[it & 1][0];

    U128 a0, a1, b0, b1;
    const bool more = (it + 1) < nk;
    if (more) {                              // uniform branch: fetch next tile
      unsigned off = (unsigned)(it + 1) * 64;  // 32 halves = 64 bytes per K-step
      a0 = *(const U128*)(pA0 + off);
      a1 = *(const U128*)(pA1 + off);
      b0 = *(const U128*)(pB0 + off);
      b1 = *(const U128*)(pB1 + off);
      __builtin_prefetch(pA0 + off + 64, 0, 3);
      __builtin_prefetch(pB0 + off + 64, 0, 3);
    }

    // A fragment: lanes 0-15 hold K{0..7,16..23}; lanes 16-31 hold K{8..15,24..31}
    const int aoff = laneHi ? 8 : 0;
    v16bf afr[2];
#pragma unroll
    for (int i = 0; i < 2; ++i) {
      int row = mW + 16 * i + laneLow;
      FragU u;
      u.q[0] = *(const U128*)(rA + row * 40 + aoff);
      u.q[1] = *(const U128*)(rA + row * 40 + aoff + 16);
      afr[i] = u.v;
    }
    // B fragment: lanes 0-15 hold K{0..7,16..23}; lanes 16-31 hold K{8..15,24..31} (col = laneLow)
    const int boff = laneHi ? 8 : 0;
    v16bf bfr[4];
#pragma unroll
    for (int j = 0; j < 4; ++j) {
      int row = nW + 16 * j + laneLow;
      FragU u;
      u.q[0] = *(const U128*)(rB + row * 40 + boff);
      u.q[1] = *(const U128*)(rB + row * 40 + boff + 16);
      bfr[j] = u.v;
    }

    if (more) {                              // stage tile it+1 into other buffer
      unsigned short* wA = &sA[(it + 1) & 1][0];
      unsigned short* wB = &sB[(it + 1) & 1][0];
      *(U128*)(wA + ar * 40 + ac)        = a0;
      *(U128*)(wA + (ar + 64) * 40 + ac) = a1;
      *(U128*)(wB + ar * 40 + ac)        = b0;
      *(U128*)(wB + (ar + 64) * 40 + ac) = b1;
    }

#pragma unroll
    for (int i = 0; i < 2; ++i)
#pragma unroll
      for (int j = 0; j < 4; ++j)
        acc[i][j] = __builtin_amdgcn_wmma_f32_16x16x32_bf16(
            false, afr[i], false, bfr[j], (short)0, acc[i][j], false, false);
  }

  // Store D: lanes 0-15 -> N=laneLow, M=vv; lanes 16-31 -> M=vv+8
#pragma unroll
  for (int i = 0; i < 2; ++i) {
    int rbase = mBlk + mW + 16 * i + laneHi * 8;
#pragma unroll
    for (int j = 0; j < 4; ++j) {
      int col = nBlk + nW + 16 * j + laneLow;
      if (col < Ncols) {
#pragma unroll
        for (int vv = 0; vv < 8; ++vv) {
          int r = rbase + vv;
          if (r < M) Cout[(size_t)r * ldC + col] = acc[i][j][vv];
        }
      }
    }
  }
}

// ---------------------------------------------------------------------------
// alpha_src / alpha_dst:  as[n,h] = dot(hlin[n, h*C : h*C+C], a_s[h])
// grid (N, H), block 128
// ---------------------------------------------------------------------------
__global__ void gat_alpha(const float* __restrict__ hlin,
                          const float* __restrict__ avs, const float* __restrict__ avd,
                          float* __restrict__ as_, float* __restrict__ ad_,
                          int H, int C, int ld) {
  int n = blockIdx.x, h = blockIdx.y;
  const float* hp = hlin + (size_t)n * ld + h * C;
  const float* sp = avs + (size_t)h * C;
  const float* dp = avd + (size_t)h * C;
  float ps = 0.f, pd = 0.f;
  for (int c = threadIdx.x; c < C; c += blockDim.x) {
    float v = hp[c];
    ps += v * sp[c];
    pd += v * dp[c];
  }
  __shared__ float rs[128], rd[128];
  rs[threadIdx.x] = ps; rd[threadIdx.x] = pd;
  __syncthreads();
  for (int o = 64; o > 0; o >>= 1) {
    if ((int)threadIdx.x < o) { rs[threadIdx.x] += rs[threadIdx.x + o]; rd[threadIdx.x] += rd[threadIdx.x + o]; }
    __syncthreads();
  }
  if (threadIdx.x == 0) { as_[n * H + h] = rs[0]; ad_[n * H + h] = rd[0]; }
}

// ---------------------------------------------------------------------------
// Edge pass 1: segment max via order-preserving atomicMax on uint
// ---------------------------------------------------------------------------
__global__ void gat_edge_max(const int* __restrict__ ei, int E, int Etot,
                             const float* __restrict__ as_, const float* __restrict__ ad_,
                             unsigned* __restrict__ mu, int H) {
  long idx = (long)blockIdx.x * blockDim.x + threadIdx.x;
  long tot = (long)Etot * H;
  if (idx >= tot) return;
  int e = (int)(idx / H), h = (int)(idx % H);
  int src = (e < E) ? ei[e] : (e - E);
  int dst = (e < E) ? ei[E + e] : (e - E);
  float v = as_[src * H + h] + ad_[dst * H + h];
  v = (v > 0.f) ? v : 0.2f * v;             // leaky_relu(., 0.2)
  atomicMax(&mu[dst * H + h], ordU(v));
}

// ---------------------------------------------------------------------------
// Edge pass 2: ee = exp(e - m[dst]); s[dst] += ee
// ---------------------------------------------------------------------------
__global__ void gat_edge_exp(const int* __restrict__ ei, int E, int Etot,
                             const float* __restrict__ as_, const float* __restrict__ ad_,
                             const unsigned* __restrict__ mu, float* __restrict__ s,
                             float* __restrict__ ee, int H) {
  long idx = (long)blockIdx.x * blockDim.x + threadIdx.x;
  long tot = (long)Etot * H;
  if (idx >= tot) return;
  int e = (int)(idx / H), h = (int)(idx % H);
  int src = (e < E) ? ei[e] : (e - E);
  int dst = (e < E) ? ei[E + e] : (e - E);
  float v = as_[src * H + h] + ad_[dst * H + h];
  v = (v > 0.f) ? v : 0.2f * v;
  float m  = unordU(mu[dst * H + h]);
  float ev = __expf(v - m);
  ee[idx] = ev;
  atomicAdd(&s[dst * H + h], ev);
}

// ---------------------------------------------------------------------------
// Edge pass 3: agg[dst,f] += ee[e,h(f)] * hlin[src,f]   (block per edge)
// ---------------------------------------------------------------------------
__global__ void gat_edge_agg(const int* __restrict__ ei, int E,
                             const float* __restrict__ hlin, const float* __restrict__ ee,
                             float* __restrict__ agg, int H, int C, int ld) {
  int e = blockIdx.x;
  int src = (e < E) ? ei[e] : (e - E);
  int dst = (e < E) ? ei[E + e] : (e - E);
  __shared__ float w[8];
  if ((int)threadIdx.x < H) w[threadIdx.x] = ee[(size_t)e * H + threadIdx.x];
  __syncthreads();
  const float* hs = hlin + (size_t)src * ld;
  float* ag = agg + (size_t)dst * ld;
  for (int h = 0; h < H; ++h) {
    float wh = w[h];
    int base = h * C;
    for (int c = threadIdx.x; c < C; c += blockDim.x)
      atomicAdd(&ag[base + c], wh * hs[base + c]);
  }
}

// ---------------------------------------------------------------------------
// Epilogue layers 1/2: v = elu(agg/s + bias) (+ skip); write bf16 (K-padded)
// ---------------------------------------------------------------------------
__global__ void gat_post_cat(const float* __restrict__ agg, const float* __restrict__ s,
                             const float* __restrict__ bias, const float* __restrict__ skip,
                             unsigned short* __restrict__ xb,
                             int N, int H, int C, int ld) {
  long idx = (long)blockIdx.x * blockDim.x + threadIdx.x;
  long tot = (long)N * ld;
  if (idx >= tot) return;
  int n = (int)(idx / ld), f = (int)(idx % ld);
  float v = 0.f;
  int HC = H * C;
  if (f < HC) {
    int h = f / C;
    v = agg[idx] / s[n * H + h] + bias[f];
    v = (v > 0.f) ? v : (__expf(v) - 1.f);  // ELU
    if (skip) v += skip[idx];
  }
  xb[idx] = f2bf(v);
}

// ---------------------------------------------------------------------------
// Final epilogue: out[n,k] = mean_h(agg[n,h*C+k]/s[n,h]) + b3[k]
// ---------------------------------------------------------------------------
__global__ void gat_post_out(const float* __restrict__ agg, const float* __restrict__ s,
                             const float* __restrict__ b3, float* __restrict__ out,
                             int N, int H, int C, int ld) {
  long idx = (long)blockIdx.x * blockDim.x + threadIdx.x;
  long tot = (long)N * C;
  if (idx >= tot) return;
  int n = (int)(idx / C), k = (int)(idx % C);
  float a = 0.f;
  for (int h = 0; h < H; ++h)
    a += agg[(size_t)n * ld + h * C + k] / s[n * H + h];
  out[idx] = a / (float)H + b3[k];
}

// ---------------------------------------------------------------------------
// Host orchestration
// ---------------------------------------------------------------------------
extern "C" void kernel_launch(void* const* d_in, const int* in_sizes, int n_in,
                              void* d_out, int out_size, void* d_ws, size_t ws_size,
                              hipStream_t stream) {
  (void)in_sizes; (void)n_in; (void)out_size; (void)ws_size;

  const float* x    = (const float*)d_in[0];
  const int*   ei   = (const int*)d_in[1];
  const float* W1   = (const float*)d_in[2];
  const float* a1s  = (const float*)d_in[3];
  const float* a1d  = (const float*)d_in[4];
  const float* b1   = (const float*)d_in[5];
  const float* W2   = (const float*)d_in[6];
  const float* a2s  = (const float*)d_in[7];
  const float* a2d  = (const float*)d_in[8];
  const float* b2   = (const float*)d_in[9];
  const float* Wsk  = (const float*)d_in[10];
  const float* W3   = (const float*)d_in[11];
  const float* a3s  = (const float*)d_in[12];
  const float* a3d  = (const float*)d_in[13];
  const float* b3   = (const float*)d_in[14];

  const int N = 10000, E = 160000, Etot = E + N;
  const int F = 50, K1 = 64;                 // layer-1 K padded 50 -> 64
  const int HC = 1400, KP = 1408;            // layers 1/2 width, padded
  const int H12 = 4, C12 = 350;
  const int H3 = 6, C3 = 121, HC3 = 726, LD3 = 736;

  char* ws = (char*)d_ws;
  size_t off = 0;
  auto alloc = [&](size_t bytes) -> char* {
    char* p = ws + off;
    off = (off + bytes + 255) & ~(size_t)255;
    return p;
  };

  unsigned short* XB1 = (unsigned short*)alloc((size_t)N * K1 * 2);
  unsigned short* W1b = (unsigned short*)alloc((size_t)HC * K1 * 2);
  unsigned short* W2b = (unsigned short*)alloc((size_t)HC * KP * 2);
  unsigned short* WSb = (unsigned short*)alloc((size_t)HC * KP * 2);
  unsigned short* W3b = (unsigned short*)alloc((size_t)HC3 * KP * 2);
  unsigned short* XB  = (unsigned short*)alloc((size_t)N * KP * 2);
  float* HLIN = (float*)alloc((size_t)N * KP * 4);
  float* AGG  = (float*)alloc((size_t)N * KP * 4);
  float* SKIP = (float*)alloc((size_t)N * KP * 4);
  float* EEb  = (float*)alloc((size_t)Etot * H3 * 4);
  float* AS   = (float*)alloc((size_t)N * H3 * 4);
  float* AD   = (float*)alloc((size_t)N * H3 * 4);
  unsigned* MU = (unsigned*)alloc((size_t)N * H3 * 4);
  float* S    = (float*)alloc((size_t)N * H3 * 4);

  const int thr = 256;
  auto nb = [&](long tot) { return (int)((tot + thr - 1) / thr); };

  // --- weight / input conversions to bf16 (K zero-padded) ---
  gat_cvt_bf16<<<nb((long)N * K1), thr, 0, stream>>>(x,   XB1, N,   F,  K1);
  gat_cvt_bf16<<<nb((long)HC * K1), thr, 0, stream>>>(W1, W1b, HC,  F,  K1);
  gat_cvt_bf16<<<nb((long)HC * KP), thr, 0, stream>>>(W2, W2b, HC,  HC, KP);
  gat_cvt_bf16<<<nb((long)HC * KP), thr, 0, stream>>>(Wsk, WSb, HC, HC, KP);
  gat_cvt_bf16<<<nb((long)HC3 * KP), thr, 0, stream>>>(W3, W3b, HC3, HC, KP);

  dim3 gN12((HC + 127) / 128, (N + 127) / 128);
  dim3 gN3((HC3 + 127) / 128, (N + 127) / 128);

  // ===== Layer 1 =====
  gat_gemm_bf16<<<gN12, 256, 0, stream>>>(XB1, W1b, HLIN, N, HC, K1, KP);
  hipMemsetAsync(AGG, 0, (size_t)N * KP * 4, stream);
  hipMemsetAsync(MU,  0, (size_t)N * H12 * 4, stream);
  hipMemsetAsync(S,   0, (size_t)N * H12 * 4, stream);
  gat_alpha<<<dim3(N, H12), 128, 0, stream>>>(HLIN, a1s, a1d, AS, AD, H12, C12, KP);
  gat_edge_max<<<nb((long)Etot * H12), thr, 0, stream>>>(ei, E, Etot, AS, AD, MU, H12);
  gat_edge_exp<<<nb((long)Etot * H12), thr, 0, stream>>>(ei, E, Etot, AS, AD, MU, S, EEb, H12);
  gat_edge_agg<<<Etot, 256, 0, stream>>>(ei, E, HLIN, EEb, AGG, H12, C12, KP);
  gat_post_cat<<<nb((long)N * KP), thr, 0, stream>>>(AGG, S, b1, nullptr, XB, N, H12, C12, KP);

  // ===== Layer 2 (+ skip GEMM from layer-1 output) =====
  gat_gemm_bf16<<<gN12, 256, 0, stream>>>(XB, W2b, HLIN, N, HC, KP, KP);
  gat_gemm_bf16<<<gN12, 256, 0, stream>>>(XB, WSb, SKIP, N, HC, KP, KP);
  hipMemsetAsync(AGG, 0, (size_t)N * KP * 4, stream);
  hipMemsetAsync(MU,  0, (size_t)N * H12 * 4, stream);
  hipMemsetAsync(S,   0, (size_t)N * H12 * 4, stream);
  gat_alpha<<<dim3(N, H12), 128, 0, stream>>>(HLIN, a2s, a2d, AS, AD, H12, C12, KP);
  gat_edge_max<<<nb((long)Etot * H12), thr, 0, stream>>>(ei, E, Etot, AS, AD, MU, H12);
  gat_edge_exp<<<nb((long)Etot * H12), thr, 0, stream>>>(ei, E, Etot, AS, AD, MU, S, EEb, H12);
  gat_edge_agg<<<Etot, 256, 0, stream>>>(ei, E, HLIN, EEb, AGG, H12, C12, KP);
  gat_post_cat<<<nb((long)N * KP), thr, 0, stream>>>(AGG, S, b2, SKIP, XB, N, H12, C12, KP);

  // ===== Layer 3 (mean over heads) =====
  gat_gemm_bf16<<<gN3, 256, 0, stream>>>(XB, W3b, HLIN, N, HC3, KP, LD3);
  hipMemsetAsync(AGG, 0, (size_t)N * LD3 * 4, stream);
  hipMemsetAsync(MU,  0, (size_t)N * H3 * 4, stream);
  hipMemsetAsync(S,   0, (size_t)N * H3 * 4, stream);
  gat_alpha<<<dim3(N, H3), 128, 0, stream>>>(HLIN, a3s, a3d, AS, AD, H3, C3, LD3);
  gat_edge_max<<<nb((long)Etot * H3), thr, 0, stream>>>(ei, E, Etot, AS, AD, MU, H3);
  gat_edge_exp<<<nb((long)Etot * H3), thr, 0, stream>>>(ei, E, Etot, AS, AD, MU, S, EEb, H3);
  gat_edge_agg<<<Etot, 256, 0, stream>>>(ei, E, HLIN, EEb, AGG, H3, C3, LD3);
  gat_post_out<<<nb((long)N * C3), thr, 0, stream>>>(AGG, S, b3, (float*)d_out, N, H3, C3, LD3);
}
